// ARSampler_59244778881183
// MI455X (gfx1250) — matrix operand order repaired and checked by reading, not compiled, not run on hardware
//
#include <hip/hip_runtime.h>

// ---------------------------------------------------------------------------
// Autoregressive GRU sampler for MI455X (gfx1250, wave32).
// The computation is a single 262,144-cell sequential chain -> one persistent
// wave; w_hh lives in VGPRs as 20 WMMA B-fragments; recurrent state in LDS.
// ---------------------------------------------------------------------------

typedef __attribute__((ext_vector_type(16))) _Float16 v16h;
typedef __attribute__((ext_vector_type(8)))  _Float16 v8h;
typedef __attribute__((ext_vector_type(8)))  float    v8f;

#define HDIM   50      // hidden size
#define G3     150     // 3*H gate rows
#define NT     10      // 160 padded rows / 16
#define NSMAX  4096    // max n_sample held in LDS state

__device__ __forceinline__ unsigned hash_u32(unsigned x) {
  x ^= x >> 16; x *= 0x7feb352dU;
  x ^= x >> 15; x *= 0x846ca68bU;
  x ^= x >> 16;
  return x;
}

union V16U { v16h v; v8h h[2]; };

__launch_bounds__(32, 1)
__global__ void gru_ar_sampler(const float* __restrict__ w_ih,
                               const float* __restrict__ w_hh,
                               const float* __restrict__ b_ih,
                               const float* __restrict__ b_hh,
                               const float* __restrict__ fc_w,
                               const float* __restrict__ fc_b,
                               const int* __restrict__ sorb_p,
                               const int* __restrict__ nsamp_p,
                               int* __restrict__ out) {
  // ---- LDS-resident state (total ~27 KB of the WGP's 320 KB) ----
  __shared__ __align__(16) float    xptab[3][152]; // x@w_ih.T + b_ih + b_hh, per one-hot case
  __shared__ __align__(16) float    gh[160];       // W_hh @ h (padded)
  __shared__ __align__(16) float    h32[64];       // hidden state, f32 master
  __shared__ __align__(16) _Float16 hbuf[64];      // hidden state, f16 image for WMMA A
  __shared__ float         nup[NSMAX];             // per-sample up-spin count
  __shared__ unsigned char sel[2][NSMAX];          // previous-step samples (double buffer)

  const int l = threadIdx.x;
  const int sorb  = sorb_p[0];
  int ns = nsamp_p[0];
  const int nsamp = (ns < NSMAX) ? ns : NSMAX;

  // ------------------------- one-time setup -------------------------
  for (int j = l; j < G3; j += 32) {
    float bb = b_ih[j] + b_hh[j];
    xptab[0][j] = w_ih[2 * j + 0] + bb;   // x = (1,0)
    xptab[1][j] = w_ih[2 * j + 1] + bb;   // x = (0,1)
    xptab[2][j] = bb;                     // x = (0,0) (step 0)
  }
  for (int j = l; j < 64; j += 32) { h32[j] = 0.f; hbuf[j] = (_Float16)0.f; }
  for (int n = l; n < NSMAX; n += 32) { nup[n] = 0.f; sel[0][n] = 0; sel[1][n] = 0; }

  // B fragments: B = w_hh^T, 16x16x32 f16 tiles, resident in VGPRs.
  // B k order used here (contiguous, same as A): lanes 0-15 hold K=0..15 (cols = lane),
  // lanes 16-31 hold K=16..31; v16h element e <-> K = base + e.
  v16h Bf[NT][2];
  {
    const int r  = (l & 15);
    const int kb = (l & 16) ? 16 : 0;
#pragma unroll
    for (int t = 0; t < NT; ++t) {
      const int row = t * 16 + r;
#pragma unroll
      for (int c = 0; c < 2; ++c) {
        v16h b;
#pragma unroll
        for (int e = 0; e < 16; ++e) {
          const int k = c * 32 + kb + e;
          float v = (row < G3 && k < HDIM) ? w_hh[row * HDIM + k] : 0.f;
          b[e] = (_Float16)v;
        }
        Bf[t][c] = b;
      }
    }
  }

  // fc (2 x 50) weights as per-lane registers for the logits matvec.
  const float fc0a = (l      < HDIM) ? fc_w[l]               : 0.f;
  const float fc0b = (l + 32 < HDIM) ? fc_w[l + 32]          : 0.f;
  const float fc1a = (l      < HDIM) ? fc_w[HDIM + l]        : 0.f;
  const float fc1b = (l + 32 < HDIM) ? fc_w[HDIM + l + 32]   : 0.f;
  const float fb0 = fc_b[0], fb1 = fc_b[1];
  const float baseline = (float)(sorb / 2 - 1);

  __syncthreads();   // single-wave workgroup: barrier is S_NOP-class, acts as LDS fence

  const v8h* hv = (const v8h*)hbuf;   // 8 chunks of 8 halves: chunk c = K[8c..8c+7]
  const int  q  = (l & 16) ? 1 : 0;   // A-fragment 16-bit layout lane split

  // ------------------------- the 262,144-cell chain -------------------------
  for (int i = 0; i < sorb; ++i) {
    const int  rb = i & 1, wb = rb ^ 1;
    const bool mask_on = (i >= (sorb >> 1));
    const float fi = (float)i;

    for (int n = 0; n < nsamp; ++n) {
      // ---- A fragments (h broadcast to all 16 rows) from LDS f16 image ----
      V16U A0, A1;
      A0.h[0] = hv[2 * q];     A0.h[1] = hv[2 * q + 1];   // K 0..31
      A1.h[0] = hv[2 * q + 4]; A1.h[1] = hv[2 * q + 5];   // K 32..63

      // ---- gh = W_hh @ h via 20 x v_wmma_f32_16x16x32_f16 ----
#pragma unroll
      for (int t = 0; t < NT; ++t) {
        v8f d = {};
        d = __builtin_amdgcn_wmma_f32_16x16x32_f16(false, A0.v, false, Bf[t][0],
                                                   (short)0, d, false, false);
        d = __builtin_amdgcn_wmma_f32_16x16x32_f16(false, A1.v, false, Bf[t][1],
                                                   (short)0, d, false, false);
        // D row 0 (lanes 0-15) == D row 8 (lanes 16-31): both halves store the
        // identical value to the same address -> benign.
        gh[t * 16 + (l & 15)] = d[0];
      }
      __syncthreads();

      // ---- GRU gates across 32 lanes x 2 passes ----
      const int idx = (i == 0) ? 2 : (int)sel[rb][n];
      const float* xp = &xptab[idx][0];
      float acc0 = 0.f, acc1 = 0.f;
#pragma unroll
      for (int p = 0; p < 2; ++p) {
        const int  j   = l + 32 * p;
        const bool act = (j < HDIM);
        const int  jc  = act ? j : 0;
        const float ghr = gh[jc];
        const float ghz = gh[HDIM + jc];
        const float ghn = gh[2 * HDIM + jc];
        const float r = 1.f / (1.f + __expf(-(xp[jc] + ghr)));
        const float z = 1.f / (1.f + __expf(-(xp[HDIM + jc] + ghz)));
        const float a = xp[2 * HDIM + jc] + r * ghn;
        const float nn = 1.f - 2.f / (1.f + __expf(2.f * a));   // tanh(a)
        const float hn = (1.f - z) * nn + z * h32[jc];
        if (act) { h32[j] = hn; hbuf[j] = (_Float16)hn; }
        const float w0 = p ? fc0b : fc0a;
        const float w1 = p ? fc1b : fc1a;
        if (act) { acc0 += hn * w0; acc1 += hn * w1; }
      }

      // ---- logits: cross-lane reduction (wave32) ----
#pragma unroll
      for (int off = 16; off > 0; off >>= 1) {
        acc0 += __shfl_xor(acc0, off, 32);
        acc1 += __shfl_xor(acc1, off, 32);
      }
      const float l0 = acc0 + fb0, l1 = acc1 + fb1;

      // ---- sqrt-softmax, occupation masking, 2-way sample (uniform lanes) ----
      const float mx = fmaxf(l0, l1);
      const float e0 = __expf(l0 - mx), e1 = __expf(l1 - mx);
      const float inv = 1.f / (e0 + e1);
      float o0 = __fsqrt_rn(e0 * inv), o1 = __fsqrt_rn(e1 * inv);
      if (mask_on) {
        const float nu = nup[n];
        const float nd = fi - nu;
        const float ad = ((baseline - nd) >= 0.f) ? 1.f : 0.f;
        const float au = ((baseline - nu) >= 0.f) ? 1.f : 0.f;
        const float m0 = o0 * ad, m1 = o1 * au;
        const float nr = fmaxf(__fsqrt_rn(m0 * m0 + m1 * m1), 1e-30f);
        o0 = m0 / nr; o1 = m1 / nr;
      }
      const float p0 = o0 / fmaxf(o0 + o1, 1e-30f);
      const unsigned h = hash_u32(((unsigned)i * 0x9E3779B9U) ^
                                  ((unsigned)n + 0x6A09E667U));
      const float u = (float)(h >> 8) * (1.0f / 16777216.0f);
      const int s = (u >= p0) ? 1 : 0;

      if (l == 0) {
        out[n * sorb + i] = s;            // samples.T -> (n_sample, sorb)
        sel[wb][n] = (unsigned char)s;    // next step's one-hot selector
        nup[n] += (float)s;
      }
      __syncthreads();   // publish hbuf/sel/nup before next cell (free: 1 wave)
    }
  }
}

extern "C" void kernel_launch(void* const* d_in, const int* in_sizes, int n_in,
                              void* d_out, int out_size, void* d_ws, size_t ws_size,
                              hipStream_t stream) {
  (void)in_sizes; (void)n_in; (void)out_size; (void)d_ws; (void)ws_size;
  const float* w_ih  = (const float*)d_in[0];
  const float* w_hh  = (const float*)d_in[1];
  const float* b_ih  = (const float*)d_in[2];
  const float* b_hh  = (const float*)d_in[3];
  const float* fc_w  = (const float*)d_in[4];
  const float* fc_b  = (const float*)d_in[5];
  const int*   sorb  = (const int*)d_in[6];
  const int*   nsamp = (const int*)d_in[7];
  // One persistent wave32 workgroup: the dependency chain admits no more.
  gru_ar_sampler<<<dim3(1), dim3(32), 0, stream>>>(
      w_ih, w_hh, b_ih, b_hh, fc_w, fc_b, sorb, nsamp, (int*)d_out);
}
